// DecomposableAttentionModel_27616639713454
// MI455X (gfx1250) — matrix operand order repaired and checked by reading, not compiled, not run on hardware
//
#include <hip/hip_runtime.h>
#include <hip/hip_bf16.h>

// ---------------------------------------------------------------------------
// Types / helpers
// ---------------------------------------------------------------------------
typedef __attribute__((ext_vector_type(16))) __bf16 v16bf;
typedef __attribute__((ext_vector_type(8)))  __bf16 v8bf;
typedef __attribute__((ext_vector_type(8)))  float  v8f;

static __device__ __forceinline__ __bf16 f2bf(float f) {
  union { unsigned short u; __bf16 b; } v;
  unsigned int x = __float_as_uint(f);
  x += 0x7fffu + ((x >> 16) & 1u);            // round-to-nearest-even
  v.u = (unsigned short)(x >> 16);
  return v.b;
}
static __device__ __forceinline__ float bf2f(__bf16 b) {
  union { __bf16 b; unsigned short u; } v;
  v.b = b;
  return __uint_as_float(((unsigned int)v.u) << 16);
}
static __device__ __forceinline__ v16bf join16(v8bf lo, v8bf hi) {
  return __builtin_shufflevector(lo, hi, 0,1,2,3,4,5,6,7,8,9,10,11,12,13,14,15);
}

// gfx1250 async global->LDS copy (ASYNCcnt), bypasses VGPRs.
// lds_addr: 32-bit LDS byte address; gptr: 64-bit global address.
static __device__ __forceinline__ void async_load_b128(unsigned lds_addr, const void* gptr) {
  asm volatile("global_load_async_to_lds_b128 %0, %1, off"
               :: "v"(lds_addr), "v"((unsigned long long)(size_t)gptr)
               : "memory");
}
static __device__ __forceinline__ void wait_asynccnt0() {
  asm volatile("s_wait_asynccnt 0" ::: "memory");
}

// ---------------------------------------------------------------------------
// WMMA GEMM:  D = act( A * op(B) + bias )
//   A: M x K bf16 row-major (lda), batched via sA
//   BT=true : B is N x K row-major (ldb)  -> D = A * B^T   (X * W^T pattern)
//   BT=false: B is K x N row-major (ldb)  -> D = A * B     (probs * values)
//   D: M x N (ldd), f32 or bf16, batched via sD
// Block: 256 threads (8 wave32), tile 128(M) x 128(N), K-step 32.
// Double-buffered LDS; A (and B in BT mode) staged with
// global_load_async_to_lds_b128; one barrier per K-step.
// M, N multiples of 128; K multiple of 32 (all shapes here comply).
// ---------------------------------------------------------------------------
#define LDSPITCH 40   // halfs per row (32 data + 8 pad)

template<bool BT, bool RELU, bool OBF16>
__global__ __launch_bounds__(256)
void gemm_wmma_k(const __bf16* __restrict__ A, const __bf16* __restrict__ B,
                 const float* __restrict__ bias, void* __restrict__ Dv,
                 int K, int lda, int ldb, int ldd,
                 long long sA, long long sB, long long sD)
{
  __shared__ alignas(16) __bf16 lsA[2][128 * LDSPITCH];
  __shared__ alignas(16) __bf16 lsB[2][128 * LDSPITCH];

  const int tid  = threadIdx.x;
  const int bz   = blockIdx.z;
  A += (long long)bz * sA;
  B += (long long)bz * sB;
  const int rowBase = blockIdx.y * 128;
  const int colBase = blockIdx.x * 128;

  const int wave  = tid >> 5;
  const int lane  = tid & 31;
  const int waveM = wave >> 1;      // 0..3 -> 32-row slab
  const int waveN = wave & 1;       // 0..1 -> 64-col slab
  const int l15   = lane & 15;
  const int hi    = lane >> 4;      // half-wave select

  v8f acc[2][4];
#pragma unroll
  for (int i = 0; i < 2; ++i)
#pragma unroll
    for (int j = 0; j < 4; ++j)
#pragma unroll
      for (int e = 0; e < 8; ++e) acc[i][j][e] = 0.0f;

  // ---- prologue: stage K-tile 0 into buffer 0 ----
#pragma unroll
  for (int i = 0; i < 2; ++i) {
    int chunk = tid + 256 * i;            // 0..511
    int r  = chunk >> 2;                  // 0..127
    int c8 = chunk & 3;                   // 0..3 -> 8-half groups
    async_load_b128((unsigned)(size_t)&lsA[0][r * LDSPITCH + c8 * 8],
                    A + (long long)(rowBase + r) * lda + c8 * 8);
    if (BT)
      async_load_b128((unsigned)(size_t)&lsB[0][r * LDSPITCH + c8 * 8],
                      B + (long long)(colBase + r) * ldb + c8 * 8);
  }
  if (!BT) {
#pragma unroll
    for (int i = 0; i < 16; ++i) {
      int e = tid + 256 * i;              // 0..4095
      int k = e >> 7;                     // 0..31
      int n = e & 127;                    // 0..127
      lsB[0][n * LDSPITCH + k] = B[(long long)k * ldb + colBase + n];
    }
  }
  wait_asynccnt0();
  __syncthreads();

  int cur = 0;
  for (int kt = 0; kt < K; kt += 32) {
    const int  nxt     = cur ^ 1;
    const bool hasNext = (kt + 32) < K;

    // ---- issue async staging for next K-tile ----
    if (hasNext) {
#pragma unroll
      for (int i = 0; i < 2; ++i) {
        int chunk = tid + 256 * i;
        int r  = chunk >> 2;
        int c8 = chunk & 3;
        async_load_b128((unsigned)(size_t)&lsA[nxt][r * LDSPITCH + c8 * 8],
                        A + (long long)(rowBase + r) * lda + (kt + 32) + c8 * 8);
        if (BT)
          async_load_b128((unsigned)(size_t)&lsB[nxt][r * LDSPITCH + c8 * 8],
                          B + (long long)(colBase + r) * ldb + (kt + 32) + c8 * 8);
      }
    }
    // AB mode: issue next B-tile loads now (transpose store after compute)
    __bf16 breg[16];
    if (!BT && hasNext) {
#pragma unroll
      for (int i = 0; i < 16; ++i) {
        int e = tid + 256 * i;
        int k = e >> 7;
        int n = e & 127;
        breg[i] = B[(long long)(kt + 32 + k) * ldb + colBase + n];
      }
    }

    // ---- fragments per ISA 16-bit A/B layouts (from buffer `cur`) ----
    v16bf af[2], bfv[4];
#pragma unroll
    for (int tm = 0; tm < 2; ++tm) {
      int row = waveM * 32 + tm * 16 + l15;
      const __bf16* p = &lsA[cur][row * LDSPITCH + hi * 8];
      v8bf lo = *(const v8bf*)p;            // K = base..base+7
      v8bf h2 = *(const v8bf*)(p + 16);     // K = base+16..base+23
      af[tm] = join16(lo, h2);
    }
#pragma unroll
    for (int tn = 0; tn < 4; ++tn) {
      int col = waveN * 64 + tn * 16 + l15;
      const __bf16* p = &lsB[cur][col * LDSPITCH + hi * 8];
      v8bf lo = *(const v8bf*)p;            // K = base..base+7
      v8bf h2 = *(const v8bf*)(p + 16);     // K = base+16..base+23
      bfv[tn] = join16(lo, h2);
    }

#pragma unroll
    for (int tm = 0; tm < 2; ++tm)
#pragma unroll
      for (int tn = 0; tn < 4; ++tn)
        acc[tm][tn] = __builtin_amdgcn_wmma_f32_16x16x32_bf16(
            false, af[tm], false, bfv[tn], (short)0, acc[tm][tn], false, false);

    // ---- finish staging next tile ----
    if (!BT && hasNext) {
#pragma unroll
      for (int i = 0; i < 16; ++i) {
        int e = tid + 256 * i;
        int k = e >> 7;
        int n = e & 127;
        lsB[nxt][n * LDSPITCH + k] = breg[i];
      }
    }
    if (hasNext) wait_asynccnt0();
    __syncthreads();
    cur = nxt;
  }

  // ---- epilogue: C layout -> lane=(N), vgpr r -> M = base + r (+8 for hi) ----
  float*  Df = (float*)Dv;
  __bf16* Db = (__bf16*)Dv;
  const long long dbase = (long long)bz * sD;
#pragma unroll
  for (int tn = 0; tn < 4; ++tn) {
    int col = colBase + waveN * 64 + tn * 16 + l15;
    float bv = bias ? bias[col] : 0.0f;
#pragma unroll
    for (int tm = 0; tm < 2; ++tm) {
      int row0 = rowBase + waveM * 32 + tm * 16 + hi * 8;
#pragma unroll
      for (int r = 0; r < 8; ++r) {
        float v = acc[tm][tn][r] + bv;
        if (RELU) v = fmaxf(v, 0.0f);
        long long off = dbase + (long long)(row0 + r) * ldd + col;
        if (OBF16) Db[off] = f2bf(v); else Df[off] = v;
      }
    }
  }
}

// ---------------------------------------------------------------------------
// f32 -> bf16 conversions
// ---------------------------------------------------------------------------
__global__ void cvt_flat_k(const float* __restrict__ in, __bf16* __restrict__ out, int n) {
  int i = blockIdx.x * 256 + threadIdx.x;
  if (i < n) out[i] = f2bf(in[i]);
}

// Convert + scatter into the right half of a concat buffer (ldo row stride)
__global__ void cvt_pad_k(const float* __restrict__ in, __bf16* __restrict__ out,
                          int rows, int cols, int ldo, int colOff) {
  int i = blockIdx.x * 256 + threadIdx.x;
  int n = rows * cols;
  if (i < n) {
    int r = i / cols, c = i - r * cols;
    out[(long long)r * ldo + colOff + c] = f2bf(in[i]);
  }
}

// ---------------------------------------------------------------------------
// Softmax over length-256 rows of e1 (f32 in, bf16 out)
// ---------------------------------------------------------------------------
__global__ void softmax_rows_k(const float* __restrict__ e, __bf16* __restrict__ out) {
  __shared__ float red[256];
  int r = blockIdx.x;          // 0 .. B*L1-1
  int t = threadIdx.x;
  float x = e[(long long)r * 256 + t];
  red[t] = x; __syncthreads();
  for (int s = 128; s > 0; s >>= 1) { if (t < s) red[t] = fmaxf(red[t], red[t + s]); __syncthreads(); }
  float m = red[0]; __syncthreads();
  float ex = __expf(x - m);
  red[t] = ex; __syncthreads();
  for (int s = 128; s > 0; s >>= 1) { if (t < s) red[t] += red[t + s]; __syncthreads(); }
  out[(long long)r * 256 + t] = f2bf(ex / red[0]);
}

// Softmax over columns: row j of e1[b]^T; out stored [B, L2, L1] contiguous
__global__ void softmax_cols_k(const float* __restrict__ e, __bf16* __restrict__ out) {
  __shared__ float red[256];
  int rb = blockIdx.x;         // b*256 + j
  int b = rb >> 8, j = rb & 255;
  int t = threadIdx.x;
  float x = e[(long long)b * 65536 + (long long)t * 256 + j];
  red[t] = x; __syncthreads();
  for (int s = 128; s > 0; s >>= 1) { if (t < s) red[t] = fmaxf(red[t], red[t + s]); __syncthreads(); }
  float m = red[0]; __syncthreads();
  float ex = __expf(x - m);
  red[t] = ex; __syncthreads();
  for (int s = 128; s > 0; s >>= 1) { if (t < s) red[t] += red[t + s]; __syncthreads(); }
  out[(long long)rb * 256 + t] = f2bf(ex / red[0]);
}

// ---------------------------------------------------------------------------
// Sum over sequence length: s[b][h] = sum_i cmp[b][i][h]  (bf16 in/out)
// ---------------------------------------------------------------------------
__global__ void sum_l_k(const __bf16* __restrict__ cmp, __bf16* __restrict__ aggcat, int colOff) {
  int b = blockIdx.x;
  int h = blockIdx.y * 256 + threadIdx.x;
  const __bf16* p = cmp + (long long)b * 256 * 1024 + h;
  float acc = 0.0f;
  for (int i = 0; i < 256; ++i) acc += bf2f(p[(long long)i * 1024]);
  aggcat[(long long)b * 2048 + colOff + h] = f2bf(acc);
}

// ---------------------------------------------------------------------------
// Final linear: out[b][o] = dot(h[b], fin_w[o]) + fin_b[o]   (OUT=3)
// ---------------------------------------------------------------------------
__global__ void final_linear_k(const __bf16* __restrict__ h, const float* __restrict__ w,
                               const float* __restrict__ b, float* __restrict__ out) {
  int o  = blockIdx.x;     // 0..2
  int bi = threadIdx.x;    // 0..127
  float acc = 0.0f;
  for (int k = 0; k < 1024; ++k)
    acc += bf2f(h[(long long)bi * 1024 + k]) * w[o * 1024 + k];
  out[bi * 3 + o] = acc + b[o];
}

// ---------------------------------------------------------------------------
// Host orchestration
// ---------------------------------------------------------------------------
extern "C" void kernel_launch(void* const* d_in, const int* in_sizes, int n_in,
                              void* d_out, int out_size, void* d_ws, size_t ws_size,
                              hipStream_t stream) {
  (void)in_sizes; (void)n_in; (void)out_size; (void)ws_size;

  const float* sent1 = (const float*)d_in[0];   // [128,256,512]
  const float* sent2 = (const float*)d_in[1];
  const float* f_w1  = (const float*)d_in[2];   // [1024,512]
  const float* f_b1  = (const float*)d_in[3];
  const float* f_w2  = (const float*)d_in[4];   // [1024,1024]
  const float* f_b2  = (const float*)d_in[5];
  const float* g_w1  = (const float*)d_in[6];   // [1024,1024]
  const float* g_b1  = (const float*)d_in[7];
  const float* g_w2  = (const float*)d_in[8];   // [1024,1024]
  const float* g_b2  = (const float*)d_in[9];
  const float* h_w1  = (const float*)d_in[10];  // [1024,2048]
  const float* h_b1  = (const float*)d_in[11];
  const float* h_w2  = (const float*)d_in[12];  // [1024,1024]
  const float* h_b2  = (const float*)d_in[13];
  const float* fin_w = (const float*)d_in[14];  // [3,1024]
  const float* fin_b = (const float*)d_in[15];

  const int B = 128, L = 256, E = 512, H = 1024;
  const long long ML = (long long)B * L;        // 32768 rows

  // ---- workspace bump allocator ----
  char* ws = (char*)d_ws;
  auto alloc = [&](size_t bytes) -> char* {
    char* p = ws; ws += (bytes + 255) & ~(size_t)255; return p;
  };
  __bf16* s1b    = (__bf16*)alloc(ML * E * 2);          // sent1 bf16
  __bf16* s2b    = (__bf16*)alloc(ML * E * 2);          // sent2 bf16
  __bf16* wf1    = (__bf16*)alloc((size_t)H * E * 2);
  __bf16* wf2    = (__bf16*)alloc((size_t)H * H * 2);
  __bf16* wg1    = (__bf16*)alloc((size_t)H * 1024 * 2);
  __bf16* wg2    = (__bf16*)alloc((size_t)H * H * 2);
  __bf16* wh1    = (__bf16*)alloc((size_t)H * 2048 * 2);
  __bf16* wh2    = (__bf16*)alloc((size_t)H * H * 2);
  __bf16* tmpH   = (__bf16*)alloc(ML * H * 2);          // hidden scratch
  __bf16* f1buf  = (__bf16*)alloc(ML * H * 2);          // f1 -> concat1 -> cmp1
  __bf16* f2buf  = (__bf16*)alloc(ML * H * 2);          // f2 -> concat2 -> cmp2
  float*  e1     = (float*) alloc(ML * (size_t)L * 4);  // [B,L1,L2] f32
  __bf16* watt   = (__bf16*)alloc(ML * (size_t)L * 2);  // w1a then w2a
  __bf16* aggcat = (__bf16*)alloc((size_t)B * 2048 * 2);
  __bf16* aggh   = (__bf16*)alloc((size_t)B * H * 2);
  __bf16* agg2   = (__bf16*)alloc((size_t)B * H * 2);

  auto blocks = [](long long n) { return (unsigned)((n + 255) / 256); };

  // ---- conversions to bf16 ----
  cvt_flat_k<<<blocks(ML * E), 256, 0, stream>>>(sent1, s1b, (int)(ML * E));
  cvt_flat_k<<<blocks(ML * E), 256, 0, stream>>>(sent2, s2b, (int)(ML * E));
  cvt_flat_k<<<blocks((long long)H * E), 256, 0, stream>>>(f_w1, wf1, H * E);
  cvt_flat_k<<<blocks((long long)H * H), 256, 0, stream>>>(f_w2, wf2, H * H);
  cvt_flat_k<<<blocks((long long)H * 1024), 256, 0, stream>>>(g_w1, wg1, H * 1024);
  cvt_flat_k<<<blocks((long long)H * H), 256, 0, stream>>>(g_w2, wg2, H * H);
  cvt_flat_k<<<blocks((long long)H * 2048), 256, 0, stream>>>(h_w1, wh1, H * 2048);
  cvt_flat_k<<<blocks((long long)H * H), 256, 0, stream>>>(h_w2, wh2, H * H);

  // ---- Attend MLP f: h = relu(X W1^T + b1); f = relu(h W2^T + b2) ----
  dim3 gBig(H / 128, (unsigned)(ML / 128), 1);
  gemm_wmma_k<true, true, true><<<gBig, 256, 0, stream>>>(
      s1b, wf1, f_b1, tmpH, E, E, E, H, 0, 0, 0);
  gemm_wmma_k<true, true, true><<<gBig, 256, 0, stream>>>(
      tmpH, wf2, f_b2, f1buf, H, H, H, H, 0, 0, 0);
  gemm_wmma_k<true, true, true><<<gBig, 256, 0, stream>>>(
      s2b, wf1, f_b1, tmpH, E, E, E, H, 0, 0, 0);
  gemm_wmma_k<true, true, true><<<gBig, 256, 0, stream>>>(
      tmpH, wf2, f_b2, f2buf, H, H, H, H, 0, 0, 0);

  // ---- Scores: e1[b] = f1[b] * f2[b]^T   [256x1024]x[256x1024]^T ----
  dim3 gE(L / 128, L / 128, B);
  gemm_wmma_k<true, false, false><<<gE, 256, 0, stream>>>(
      f1buf, f2buf, nullptr, e1, H, H, H, L,
      (long long)L * H, (long long)L * H, (long long)L * L);

  // ---- w1a = softmax_rows(e1); att1 = w1a * sent2 -> concat1 left ----
  softmax_rows_k<<<(unsigned)ML, 256, 0, stream>>>(e1, watt);
  dim3 gAtt(E / 128, L / 128, B);
  gemm_wmma_k<false, false, true><<<gAtt, 256, 0, stream>>>(
      watt, s2b, nullptr, f1buf, L, L, E, 2 * E,
      (long long)L * L, (long long)L * E, (long long)L * 2 * E);
  cvt_pad_k<<<blocks(ML * E), 256, 0, stream>>>(sent1, f1buf, (int)ML, E, 2 * E, E);

  // ---- w2a = softmax_cols(e1); att2 = w2a * sent1 -> concat2 left ----
  softmax_cols_k<<<(unsigned)ML, 256, 0, stream>>>(e1, watt);
  gemm_wmma_k<false, false, true><<<gAtt, 256, 0, stream>>>(
      watt, s1b, nullptr, f2buf, L, L, E, 2 * E,
      (long long)L * L, (long long)L * E, (long long)L * 2 * E);
  cvt_pad_k<<<blocks(ML * E), 256, 0, stream>>>(sent2, f2buf, (int)ML, E, 2 * E, E);

  // ---- Compare MLP g on concat buffers ----
  gemm_wmma_k<true, true, true><<<gBig, 256, 0, stream>>>(
      f1buf, wg1, g_b1, tmpH, 1024, 1024, 1024, H, 0, 0, 0);
  gemm_wmma_k<true, true, true><<<gBig, 256, 0, stream>>>(
      tmpH, wg2, g_b2, f1buf, H, H, H, H, 0, 0, 0);
  gemm_wmma_k<true, true, true><<<gBig, 256, 0, stream>>>(
      f2buf, wg1, g_b1, tmpH, 1024, 1024, 1024, H, 0, 0, 0);
  gemm_wmma_k<true, true, true><<<gBig, 256, 0, stream>>>(
      tmpH, wg2, g_b2, f2buf, H, H, H, H, 0, 0, 0);

  // ---- Aggregate: sum over L, concat [s1,s2], MLP h, final linear ----
  dim3 gSum(B, H / 256, 1);
  sum_l_k<<<gSum, 256, 0, stream>>>(f1buf, aggcat, 0);
  sum_l_k<<<gSum, 256, 0, stream>>>(f2buf, aggcat, H);

  dim3 gAgg(H / 128, 1, 1);
  gemm_wmma_k<true, true, true><<<gAgg, 256, 0, stream>>>(
      aggcat, wh1, h_b1, aggh, 2048, 2048, 2048, H, 0, 0, 0);
  gemm_wmma_k<true, true, true><<<gAgg, 256, 0, stream>>>(
      aggh, wh2, h_b2, agg2, H, H, H, H, 0, 0, 0);

  final_linear_k<<<3, 128, 0, stream>>>(agg2, fin_w, fin_b, (float*)d_out);
}
